// MobileAttention3D_27943057228269
// MI455X (gfx1250) — matrix operand order repaired and checked by reading, not compiled, not run on hardware
//
#include <hip/hip_runtime.h>
#include <hip/hip_bf16.h>

// MobileAttention3D on gfx1250 (MI455X): bf16 WMMA everywhere; activation
// layouts interleaved in channel-pair groups of 16 so BOTH A and B WMMA
// fragments load as global_load_b128. 2x2 register blocking in big GEMMs.
// NH=8, KD=64, VD=64, C=256, B=2, D=32, HW=1024, P=D*HW=32768.

typedef __attribute__((ext_vector_type(16))) __bf16 v16bf;
typedef __attribute__((ext_vector_type(8)))  float  v8f;
typedef unsigned short u16;
typedef unsigned int   u32;

union Frag { v16bf v; u16 u[16]; u32 d[8]; };

__device__ __forceinline__ u16 f2bf(float f) {
  u32 u = __float_as_uint(f);
  u32 r = u + 0x7FFFu + ((u >> 16) & 1u);   // round-to-nearest-even
  return (u16)(r >> 16);
}

__device__ __forceinline__ v8f wmma_bf16(const Frag& a, const Frag& b, v8f c) {
  return __builtin_amdgcn_wmma_f32_16x16x32_bf16(false, a.v, false, b.v, (short)0, c, false, false);
}

// K base (even) for vgpr-pair j, contiguous order k = 16*half + 2*j.
__device__ __forceinline__ int kA(int j, int half) {     // A-matrix 16x32
  return 2 * j + half * 16;
}

// ---------------- K0: weight conversion + score-accumulator zeroing ----------
__global__ void prep_kernel(const float* __restrict__ q_w, const float* __restrict__ kv_w,
                            const float* __restrict__ proj_w,
                            u16* __restrict__ qkv_w_bf, u16* __restrict__ proj_w_bf,
                            float* __restrict__ s_ws) {
  int i = blockIdx.x * blockDim.x + threadIdx.x;
  if (i < 640 * 256) {
    float v = (i < 512 * 256) ? q_w[i] : kv_w[i - 512 * 256];
    qkv_w_bf[i] = f2bf(v);
  }
  if (i < 256 * 512) proj_w_bf[i] = f2bf(proj_w[i]);
  if (i < 2 * 8 * 32 * 32) s_ws[i] = 0.0f;
}

// ---------------- K0b: pack x fp32 -> bf16, group-of-16-pair interleaved -----
// x_t[b][cg][p][ip] (u32) = { bf16(x[cg*32+2ip, p]), bf16(x[cg*32+2ip+1, p]) }
__global__ __launch_bounds__(256) void xpack(const float* __restrict__ x, u32* __restrict__ xt) {
  size_t i = (size_t)blockIdx.x * 256 + threadIdx.x;   // B*8*32768 = 524288
  int p  = (int)(i & 32767);
  int t  = (int)(i >> 15);
  int cg = t & 7;
  int b  = t >> 3;
  const float* xb = x + ((size_t)b * 256 + cg * 32) * 32768 + p;
  u32* dst = xt + ((size_t)(b * 8 + cg) * 32768 + p) * 16;
  u32 vals[16];
#pragma unroll
  for (int ip = 0; ip < 16; ++ip) {
    u32 lo = f2bf(xb[(size_t)(2 * ip)     * 32768]);
    u32 hi = f2bf(xb[(size_t)(2 * ip + 1) * 32768]);
    vals[ip] = lo | (hi << 16);
  }
#pragma unroll
  for (int ip = 0; ip < 16; ++ip) dst[ip] = vals[ip];   // 4x b128, contiguous
}

// ---------------- K1: fused q/kv 1x1 conv  (640x256)x(256x32768) per batch --
// 2x2 blocking: each wave computes 32(o) x 32(p); all frag loads are b128.
__global__ __launch_bounds__(256) void qkv_gemm(const u32* __restrict__ xt,
    const u16* __restrict__ wbf, const float* __restrict__ q_b, const float* __restrict__ kv_b,
    u16* __restrict__ q_ws, u16* __restrict__ k_ws, u16* __restrict__ v_ws) {
  const int wave = (blockIdx.x * blockDim.x + threadIdx.x) >> 5;
  const int lane = threadIdx.x & 31;
  const int half = lane >> 4, lm = lane & 15;
  int pt = wave % 1024; int t = wave / 1024;
  int ot = t % 20;      int b = t / 20;
  const int obase = ot * 32, pbase = pt * 32;
  const u32* xb = xt + (size_t)b * 8 * 32768 * 16;
  const u16* wrow0 = wbf + (size_t)(obase + lm) * 256;
  const u16* wrow1 = wbf + (size_t)(obase + 16 + lm) * 256;

  v8f a00 = {}, a01 = {}, a10 = {}, a11 = {};
  for (int kc = 0; kc < 256; kc += 32) {
    const u32* xg  = xb + (size_t)(kc >> 5) * (32768 * 16);
    const u32* xl0 = xg + (size_t)(pbase + lm)      * 16 + half * 8;
    const u32* xl1 = xg + (size_t)(pbase + 16 + lm) * 16 + half * 8;
    Frag fa0, fa1, fb0, fb1;
#pragma unroll
    for (int j = 0; j < 8; ++j) {
      int ka = kc + kA(j, half);
      fa0.d[j] = *(const u32*)(wrow0 + ka);
      fa1.d[j] = *(const u32*)(wrow1 + ka);
      fb0.d[j] = xl0[j];                 // 8 consecutive u32 -> 2x b128
      fb1.d[j] = xl1[j];
    }
    a00 = wmma_bf16(fa0, fb0, a00);
    a01 = wmma_bf16(fa0, fb1, a01);
    a10 = wmma_bf16(fa1, fb0, a10);
    a11 = wmma_bf16(fa1, fb1, a11);
  }

  // epilogue: region is wave-uniform (o-tiles never straddle 512/576)
  if (obase < 512) {                      // q rows
    u16* qb_ = q_ws + (size_t)b * 512 * 32768;
#pragma unroll
    for (int r = 0; r < 8; ++r) {
      int o0 = obase + r + half * 8, o1 = o0 + 16;
      qb_[(size_t)o0 * 32768 + pbase + lm]      = f2bf(a00[r] + q_b[o0]);
      qb_[(size_t)o0 * 32768 + pbase + 16 + lm] = f2bf(a01[r] + q_b[o0]);
      qb_[(size_t)o1 * 32768 + pbase + lm]      = f2bf(a10[r] + q_b[o1]);
      qb_[(size_t)o1 * 32768 + pbase + 16 + lm] = f2bf(a11[r] + q_b[o1]);
    }
  } else if (obase < 576) {               // k rows
    u16* kb_ = k_ws + (size_t)b * 64 * 32768;
#pragma unroll
    for (int r = 0; r < 8; ++r) {
      int o0 = obase + r + half * 8, o1 = o0 + 16;
      kb_[(size_t)(o0 - 512) * 32768 + pbase + lm]      = f2bf(a00[r] + kv_b[o0 - 512]);
      kb_[(size_t)(o0 - 512) * 32768 + pbase + 16 + lm] = f2bf(a01[r] + kv_b[o0 - 512]);
      kb_[(size_t)(o1 - 512) * 32768 + pbase + lm]      = f2bf(a10[r] + kv_b[o1 - 512]);
      kb_[(size_t)(o1 - 512) * 32768 + pbase + 16 + lm] = f2bf(a11[r] + kv_b[o1 - 512]);
    }
  } else {                                // v rows: d-pair interleaved [vd][d/2][hw][2]
    u16* vb_ = v_ws + (size_t)b * 64 * 32768;
    int d  = pbase >> 10;                 // uniform: 32-wide p tile stays in one d
    int hw0 = pbase & 1023;
    size_t doff = (size_t)(d >> 1) * 2048 + (d & 1);
#pragma unroll
    for (int r = 0; r < 8; ++r) {
      int o0 = obase + r + half * 8, o1 = o0 + 16;
      vb_[(size_t)(o0 - 576) * 32768 + doff + (hw0 + lm) * 2]      = f2bf(a00[r] + kv_b[o0 - 512]);
      vb_[(size_t)(o0 - 576) * 32768 + doff + (hw0 + 16 + lm) * 2] = f2bf(a01[r] + kv_b[o0 - 512]);
      vb_[(size_t)(o1 - 576) * 32768 + doff + (hw0 + lm) * 2]      = f2bf(a10[r] + kv_b[o1 - 512]);
      vb_[(size_t)(o1 - 576) * 32768 + doff + (hw0 + 16 + lm) * 2] = f2bf(a11[r] + kv_b[o1 - 512]);
    }
  }
}

// ---------------- K2: scores s[dq,dk] = sum_{kd,hw} q*k  (K-split + atomics) -
__global__ __launch_bounds__(32) void scores_partial(const u16* __restrict__ q_ws,
    const u16* __restrict__ k_ws, float* __restrict__ s_ws) {
  int blk = blockIdx.x;                 // b*8*16 single-wave blocks
  int slice = blk % 16; int t = blk / 16;
  int n = t % 8;        int b = t / 8;
  int lane = threadIdx.x; int half = lane >> 4, lm = lane & 15;
  const u16* qb = q_ws + ((size_t)b * 512 + n * 64) * 32768;  // [kd][dq][hw]
  const u16* kb = k_ws + (size_t)b * 64 * 32768;              // [kd][dk][hw]

  v8f a00 = {}, a01 = {}, a10 = {}, a11 = {};
  for (int kd = slice * 4; kd < slice * 4 + 4; ++kd) {
    const u16* qch = qb + (size_t)kd * 32768;
    const u16* kch = kb + (size_t)kd * 32768;
    for (int hw0 = 0; hw0 < 1024; hw0 += 32) {
      Frag fa0, fa1, fb0, fb1;
#pragma unroll
      for (int j = 0; j < 8; ++j) {
        int ka = hw0 + kA(j, half);     // contiguous pairs -> b128 merge
        fa0.d[j] = *(const u32*)(qch + (size_t)lm        * 1024 + ka);
        fa1.d[j] = *(const u32*)(qch + (size_t)(16 + lm) * 1024 + ka);
        int kk = hw0 + 2 * j + half * 16;
        fb0.d[j] = *(const u32*)(kch + (size_t)lm        * 1024 + kk);
        fb1.d[j] = *(const u32*)(kch + (size_t)(16 + lm) * 1024 + kk);
      }
      a00 = wmma_bf16(fa0, fb0, a00);
      a01 = wmma_bf16(fa0, fb1, a01);
      a10 = wmma_bf16(fa1, fb0, a10);
      a11 = wmma_bf16(fa1, fb1, a11);
    }
  }
  float* s = s_ws + (size_t)(b * 8 + n) * 32 * 32;
#pragma unroll
  for (int r = 0; r < 8; ++r) {
    int dq = r + half * 8;
    atomicAdd(&s[(size_t)dq        * 32 + lm],      a00[r]);
    atomicAdd(&s[(size_t)dq        * 32 + 16 + lm], a01[r]);
    atomicAdd(&s[(size_t)(16 + dq) * 32 + lm],      a10[r]);
    atomicAdd(&s[(size_t)(16 + dq) * 32 + 16 + lm], a11[r]);
  }
}

// ---------------- K2b: softmax over 32-wide rows (wave32 shuffles) ----------
__global__ __launch_bounds__(32) void softmax_rows(const float* __restrict__ s_ws,
                                                   u16* __restrict__ attn_bf) {
  int row = blockIdx.x;                 // B*NH*32 = 512 rows
  int lane = threadIdx.x;
  float v = s_ws[(size_t)row * 32 + lane] * 0.125f;   // scale = VD^-0.5
  float m = v;
  for (int off = 16; off; off >>= 1) m = fmaxf(m, __shfl_xor(m, off));
  float e = __expf(v - m);
  float s = e;
  for (int off = 16; off; off >>= 1) s += __shfl_xor(s, off);
  attn_bf[(size_t)row * 32 + lane] = f2bf(e / s);
}

// ---------------- K3: o = attn @ v, two vd channels per wave ----------------
// v:  u32 view [vd][dk/2][hw]      (d-pair interleaved by K1)
// o:  u32 view [og][p][16]         (group-of-16-pair interleaved for K4 b128)
__global__ __launch_bounds__(256) void attn_v(const u16* __restrict__ attn_bf,
    const u32* __restrict__ v32, u32* __restrict__ ot) {
  int wave = (blockIdx.x * blockDim.x + threadIdx.x) >> 5;   // B*8*32*64 = 32768
  int lane = threadIdx.x & 31; int half = lane >> 4, lm = lane & 15;
  int ht = wave & 63;  int t = wave >> 6;
  int vdp = t & 31;    t >>= 5;
  int n = t & 7;       int b = t >> 3;
  const u16* arow = attn_bf + (size_t)(b * 8 + n) * 1024;             // 32x32
  const u32* vch0 = v32 + ((size_t)b * 64 + vdp * 2) * 16384;         // [dk/2][hw]
  const u32* vch1 = vch0 + 16384;
  int hwbase = ht * 16;

  Frag fa0, fa1, fb0, fb1;
#pragma unroll
  for (int j = 0; j < 8; ++j) {
    int ka = kA(j, half);               // contiguous pairs -> b128 merge
    fa0.d[j] = *(const u32*)(arow + (size_t)lm        * 32 + ka);
    fa1.d[j] = *(const u32*)(arow + (size_t)(16 + lm) * 32 + ka);
    int kb = 2 * j + half * 16;         // dk, even
    fb0.d[j] = vch0[(size_t)(kb >> 1) * 1024 + hwbase + lm];
    fb1.d[j] = vch1[(size_t)(kb >> 1) * 1024 + hwbase + lm];
  }
  v8f a00 = {}, a01 = {}, a10 = {}, a11 = {};
  a00 = wmma_bf16(fa0, fb0, a00);       // Mtile0, ch even
  a01 = wmma_bf16(fa0, fb1, a01);       // Mtile0, ch odd
  a10 = wmma_bf16(fa1, fb0, a10);       // Mtile1, ch even
  a11 = wmma_bf16(fa1, fb1, a11);       // Mtile1, ch odd

  int ocp = n * 32 + vdp;
  u32* och = ot + ((size_t)b * 16 + (ocp >> 4)) * 32768 * 16 + (ocp & 15);
#pragma unroll
  for (int r = 0; r < 8; ++r) {
    int dq = r + half * 8;
    och[(size_t)(dq        * 1024 + hwbase + lm) * 16] = (u32)f2bf(a00[r]) | ((u32)f2bf(a01[r]) << 16);
    och[(size_t)((16 + dq) * 1024 + hwbase + lm) * 16] = (u32)f2bf(a10[r]) | ((u32)f2bf(a11[r]) << 16);
  }
}

// ---------------- K4: proj conv (256x512)x(512x32768), + bias, * layer_scale
// 2x2 blocking; all frag loads are b128 thanks to o_t interleave.
__global__ __launch_bounds__(256) void proj_gemm(const u32* __restrict__ ot,
    const u16* __restrict__ pw_bf, const float* __restrict__ proj_b,
    const float* __restrict__ ls, float* __restrict__ out) {
  int wave = (blockIdx.x * blockDim.x + threadIdx.x) >> 5;   // B*8*1024 = 16384
  int lane = threadIdx.x & 31; int half = lane >> 4, lm = lane & 15;
  int pt = wave & 1023; int t = wave >> 10;
  int ct = t & 7;       int b = t >> 3;
  int cbase = ct * 32, pbase = pt * 32;
  const u32* ob = ot + (size_t)b * 16 * 32768 * 16;
  const u16* wrow0 = pw_bf + (size_t)(cbase + lm) * 512;
  const u16* wrow1 = pw_bf + (size_t)(cbase + 16 + lm) * 512;

  v8f a00 = {}, a01 = {}, a10 = {}, a11 = {};
  for (int kc = 0; kc < 512; kc += 32) {
    const u32* og  = ob + (size_t)(kc >> 5) * (32768 * 16);
    const u32* ol0 = og + (size_t)(pbase + lm)      * 16 + half * 8;
    const u32* ol1 = og + (size_t)(pbase + 16 + lm) * 16 + half * 8;
    Frag fa0, fa1, fb0, fb1;
#pragma unroll
    for (int j = 0; j < 8; ++j) {
      int ka = kc + kA(j, half);
      fa0.d[j] = *(const u32*)(wrow0 + ka);
      fa1.d[j] = *(const u32*)(wrow1 + ka);
      fb0.d[j] = ol0[j];                 // 8 consecutive u32 -> 2x b128
      fb1.d[j] = ol1[j];
    }
    a00 = wmma_bf16(fa0, fb0, a00);
    a01 = wmma_bf16(fa0, fb1, a01);
    a10 = wmma_bf16(fa1, fb0, a10);
    a11 = wmma_bf16(fa1, fb1, a11);
  }
#pragma unroll
  for (int r = 0; r < 8; ++r) {
    int c0 = cbase + r + half * 8;
    int c1 = c0 + 16;
    int p0 = pbase + lm;
    out[((size_t)b * 256 + c0) * 32768 + p0]      = (a00[r] + proj_b[c0]) * ls[c0];
    out[((size_t)b * 256 + c0) * 32768 + p0 + 16] = (a01[r] + proj_b[c0]) * ls[c0];
    out[((size_t)b * 256 + c1) * 32768 + p0]      = (a10[r] + proj_b[c1]) * ls[c1];
    out[((size_t)b * 256 + c1) * 32768 + p0 + 16] = (a11[r] + proj_b[c1]) * ls[c1];
  }
}

extern "C" void kernel_launch(void* const* d_in, const int* in_sizes, int n_in,
                              void* d_out, int out_size, void* d_ws, size_t ws_size,
                              hipStream_t stream) {
  const float* x      = (const float*)d_in[0];
  const float* q_w    = (const float*)d_in[1];
  const float* q_b    = (const float*)d_in[2];
  const float* kv_w   = (const float*)d_in[3];
  const float* kv_b   = (const float*)d_in[4];
  const float* proj_w = (const float*)d_in[5];
  const float* proj_b = (const float*)d_in[6];
  const float* ls     = (const float*)d_in[7];

  char* ws = (char*)d_ws;
  size_t off = 0;
  auto alloc = [&](size_t bytes) -> void* {
    void* p = ws + off;
    off = (off + bytes + 255) & ~(size_t)255;
    return p;
  };
  u16*   qkv_w_bf  = (u16*)alloc((size_t)640 * 256 * 2);
  u16*   proj_w_bf = (u16*)alloc((size_t)256 * 512 * 2);
  u32*   x_t       = (u32*)alloc((size_t)2 * 8 * 32768 * 16 * 4);    // interleaved bf16 pairs
  u16*   q_ws      = (u16*)alloc((size_t)2 * 512 * 32768 * 2);
  u16*   k_ws      = (u16*)alloc((size_t)2 * 64 * 32768 * 2);
  u16*   v_ws      = (u16*)alloc((size_t)2 * 64 * 32768 * 2);        // d-pair interleaved
  float* s_ws      = (float*)alloc((size_t)2 * 8 * 32 * 32 * 4);
  u16*   attn_bf   = (u16*)alloc((size_t)2 * 8 * 32 * 32 * 2);
  u32*   o_t       = (u32*)alloc((size_t)2 * 16 * 32768 * 16 * 4);   // interleaved bf16 pairs

  prep_kernel<<<640, 256, 0, stream>>>(q_w, kv_w, proj_w, qkv_w_bf, proj_w_bf, s_ws);
  xpack<<<2048, 256, 0, stream>>>(x, x_t);
  qkv_gemm<<<5120, 256, 0, stream>>>(x_t, qkv_w_bf, q_b, kv_b, q_ws, k_ws, v_ws);
  scores_partial<<<256, 32, 0, stream>>>(q_ws, k_ws, s_ws);
  softmax_rows<<<512, 32, 0, stream>>>(s_ws, attn_bf);
  attn_v<<<4096, 256, 0, stream>>>(attn_bf, (const u32*)v_ws, o_t);
  proj_gemm<<<2048, 256, 0, stream>>>(o_t, proj_w_bf, proj_b, ls, (float*)d_out);
}
